// AttnBlock_42992622633389
// MI455X (gfx1250) — compile-verified, matrix operand order checked by reading, not run on hardware
//
#include <hip/hip_runtime.h>
#include <stdint.h>

#define DI __device__ __forceinline__

typedef __attribute__((ext_vector_type(16))) _Float16 v16h;
typedef __attribute__((ext_vector_type(8)))  _Float16 h8;   // 16 bytes
typedef __attribute__((ext_vector_type(8)))  float    v8f;
typedef __attribute__((ext_vector_type(4)))  float    f32x4;
typedef __attribute__((ext_vector_type(4)))  unsigned int u32x4;
typedef __attribute__((ext_vector_type(4)))  int      i32x4;
typedef __attribute__((ext_vector_type(8)))  int      i32x8;

constexpr int Bb = 4;
constexpr int Cc = 512;
constexpr int Ll = 4096;
constexpr int Mrows = Bb * Ll; // 16384

// Tensor Data Mover availability (device pass only; host pass takes fallback).
#if defined(__gfx1250__)
#if __has_builtin(__builtin_amdgcn_tensor_load_to_lds) && \
    __has_builtin(__builtin_amdgcn_s_wait_tensorcnt)
#define HAVE_TDM 1
#endif
#endif
#ifndef HAVE_TDM
#define HAVE_TDM 0
#endif

union AFrag { v16h v; h8 h[2]; };

DI v8f zero8() { v8f z; for (int i = 0; i < 8; ++i) z[i] = 0.f; return z; }

// Load one 16x32 f16 WMMA fragment (A-operand, or B-operand with "row" = n).
// Per-lane layout (ISA 7.12.2): lane = hi*16 + r; elems e<8 -> k = hi*8+e,
// elems e>=8 -> k = 16 + hi*8 + (e-8). Two 16-byte loads per lane.
DI v16h load_frag(const _Float16* base, int stride) {
  const int lane = threadIdx.x & 31;
  const int r = lane & 15;
  const int hi = lane >> 4;
  const _Float16* p = base + (size_t)r * stride + hi * 8;
  AFrag f;
  f.h[0] = *(const h8*)(p);
  f.h[1] = *(const h8*)(p + 16);
  return f.v;
}

DI v8f wmma_f16(v16h a, v16h b, v8f c) {
  return __builtin_amdgcn_wmma_f32_16x16x32_f16(false, a, false, b, (short)0, c,
                                                false, false);
}

DI float selu(float y) {
  constexpr float kLambda = 1.0507009873554805f;
  constexpr float kAlpha  = 1.6732632423543772f;
  return (y > 0.f) ? kLambda * y : kLambda * kAlpha * (__expf(y) - 1.f);
}

#if HAVE_TDM
// Issue one TDM 2D tile load: tile_d1 rows x tile_d0 f16 elements, row stride
// `stride0` (f16 units), written contiguously row-major at LDS offset lds_addr.
// D# per ISA ch.8: group0 = {count, lds_addr, global_addr, type=2},
// group1 = {data_size=2B, tensor_dim0/1, tile_dim0/1, tensor_dim0_stride}.
DI void tdm_load_2d(unsigned lds_addr, const void* gaddr, unsigned tensor_d1,
                    unsigned tile_d0, unsigned tile_d1, unsigned stride0) {
  const unsigned long long ga = (unsigned long long)(uintptr_t)gaddr;
  u32x4 g0;
  g0[0] = 1u;                                            // count=1, user D#
  g0[1] = lds_addr;                                      // LDS byte address
  g0[2] = (unsigned)ga;                                  // global_addr[31:0]
  g0[3] = (unsigned)((ga >> 32) & 0x01FFFFFFull) | (2u << 30); // addr[56:32]|type=2
  i32x8 g1;
  g1[0] = 0x10000;                                       // data_size=1 (2 bytes)
  g1[1] = (int)((512u & 0xffffu) << 16);                 // tensor_dim0 = 512 (lo)
  g1[2] = (int)((512u >> 16) | ((tensor_d1 & 0xffffu) << 16));
  g1[3] = (int)(((tensor_d1 >> 16) & 0xffffu) | (tile_d0 << 16));
  g1[4] = (int)(tile_d1 & 0xffffu);                      // tile_dim1, tile_dim2=0
  g1[5] = (int)stride0;                                  // tensor_dim0_stride lo
  g1[6] = 0;
  g1[7] = 0;
  i32x4 z4 = {0, 0, 0, 0};
#if __clang_major__ >= 23
  i32x8 z8 = {0, 0, 0, 0, 0, 0, 0, 0};
  __builtin_amdgcn_tensor_load_to_lds(g0, g1, z4, z4, z8, 0);
#else
  __builtin_amdgcn_tensor_load_to_lds(g0, g1, z4, z4, 0);
#endif
}
#endif

// ---------------------------------------------------------------------------
// f32 -> f16 weight conversion
// ---------------------------------------------------------------------------
__global__ void cvt_kernel(const float* __restrict__ s, _Float16* __restrict__ d, int n) {
  int i = blockIdx.x * 256 + threadIdx.x;
  if (i < n) d[i] = (_Float16)s[i];
}

// ---------------------------------------------------------------------------
// LayerNorm over channels; x is [B,C,L], output h is [B*L, C] f16.
// Block = 32 consecutive l of one batch. Reads coalesced along l; writes
// staged through an LDS transpose so f16 stores are 128-bit coalesced.
// ---------------------------------------------------------------------------
__global__ __launch_bounds__(256, 1)
void ln_kernel(const float* __restrict__ x, const float* __restrict__ lw,
               const float* __restrict__ lb, _Float16* __restrict__ H) {
  __shared__ float redS[8][32];
  __shared__ float redQ[8][32];
  __shared__ float muS[32], rinvS[32];
  __shared__ _Float16 hl[32][Cc];   // 32 KB transpose tile

  const int tid = threadIdx.x;
  const int w = tid >> 5, li = tid & 31;
  const int b = blockIdx.x >> 7;            // / (L/32)
  const int l0 = (blockIdx.x & 127) * 32;

  const float* xb = x + (size_t)b * Cc * Ll + l0 + li;
  float vals[64];
  float s = 0.f, s2 = 0.f;
#pragma unroll
  for (int j = 0; j < 64; ++j) {
    const int c = w + j * 8;                 // uniform per wave -> good for SMEM
    const float v = xb[(size_t)c * Ll];      // lanes -> consecutive l: coalesced
    vals[j] = v;
    s += v;
    s2 += v * v;
  }
  redS[w][li] = s;
  redQ[w][li] = s2;
  __syncthreads();
  if (tid < 32) {
    float ts = 0.f, tq = 0.f;
    for (int ww = 0; ww < 8; ++ww) { ts += redS[ww][tid]; tq += redQ[ww][tid]; }
    const float mu = ts * (1.f / Cc);
    const float var = tq * (1.f / Cc) - mu * mu;
    muS[tid] = mu;
    rinvS[tid] = rsqrtf(var + 1e-5f);
  }
  __syncthreads();
  const float mu = muS[li], rinv = rinvS[li];
#pragma unroll
  for (int j = 0; j < 64; ++j) {
    const int c = w + j * 8;
    hl[li][c] = (_Float16)((vals[j] - mu) * rinv * lw[c] + lb[c]);
  }
  __syncthreads();
  {
    const int row = tid >> 3;                // 32 rows x 8 threads
    const int ch = (tid & 7) * 64;
    _Float16* dst = H + (size_t)(b * Ll + l0 + row) * Cc + ch;
    const _Float16* srcp = &hl[row][ch];
#pragma unroll
    for (int j = 0; j < 8; ++j)
      *(h8*)(dst + j * 8) = *(const h8*)(srcp + j * 8);
  }
}

// ---------------------------------------------------------------------------
// WMMA GEMM: Y[M,N] = A[M,K] * W[N,K]^T + bias, M=16384, N=K=512.
// MODE 0: Y -> f16 via LDS-transposed coalesced stores (QKV).
// MODE 1: bias + SELU + residual, vectorized f32 [B,C,L] out (out-proj).
// Block tile 128x64, BK=32; 8 waves = 4x2 grid of 32x32 wave tiles.
// Tiles staged to LDS by the Tensor Data Mover when available.
// ---------------------------------------------------------------------------
template <int MODE>
__global__ __launch_bounds__(256, 1)
void gemm_kernel(const _Float16* __restrict__ A,
                 const _Float16* __restrict__ Wt,   // [N][K] f16
                 const float* __restrict__ bias,
                 _Float16* __restrict__ Yh,
                 const float* __restrict__ Xres,
                 float* __restrict__ Yf) {
  constexpr int BM = 128, BN = 64, BK = 32;
  __shared__ _Float16 As[BM][BK];
  __shared__ _Float16 Bs[BN][BK];
  __shared__ _Float16 Ot[(MODE == 0) ? BM : 1][BN];  // epilogue transpose tile

  const int m0 = blockIdx.y * BM;
  const int n0 = blockIdx.x * BN;
  const int tid = threadIdx.x;
  const int w = tid >> 5;
  const int lane = tid & 31;
  const int hi = lane >> 4;
  const int ln = lane & 15;
  const int wm = w >> 1;   // 0..3
  const int wn = w & 1;    // 0..1

  v8f acc[2][2];
  for (int mt = 0; mt < 2; ++mt)
    for (int nt = 0; nt < 2; ++nt)
      acc[mt][nt] = zero8();

#if HAVE_TDM
  const unsigned asoff = (unsigned)(uintptr_t)&As[0][0];
  const unsigned bsoff = (unsigned)(uintptr_t)&Bs[0][0];
#endif

  for (int k0 = 0; k0 < Cc; k0 += BK) {
#if HAVE_TDM
    if (w == 0) {
      tdm_load_2d(asoff, A + (size_t)m0 * Cc + k0, (unsigned)Mrows, BK, BM, Cc);
      tdm_load_2d(bsoff, Wt + (size_t)n0 * Cc + k0, (unsigned)Cc, BK, BN, Cc);
      __builtin_amdgcn_s_wait_tensorcnt(0);
    }
#else
    {
      const int arow = tid >> 1, ahalf = tid & 1;
      const int brow = tid >> 2, bq = tid & 3;
      const _Float16* asrc = A + (size_t)(m0 + arow) * Cc + k0 + ahalf * 16;
      *(h8*)&As[arow][ahalf * 16] = *(const h8*)(asrc);
      *(h8*)&As[arow][ahalf * 16 + 8] = *(const h8*)(asrc + 8);
      *(h8*)&Bs[brow][bq * 8] =
          *(const h8*)(Wt + (size_t)(n0 + brow) * Cc + k0 + bq * 8);
    }
#endif
    __syncthreads();

    v16h af[2], bf[2];
    for (int mt = 0; mt < 2; ++mt) af[mt] = load_frag(&As[wm * 32 + mt * 16][0], BK);
    for (int nt = 0; nt < 2; ++nt) bf[nt] = load_frag(&Bs[wn * 32 + nt * 16][0], BK);
    for (int mt = 0; mt < 2; ++mt)
      for (int nt = 0; nt < 2; ++nt)
        acc[mt][nt] = wmma_f16(af[mt], bf[nt], acc[mt][nt]);
    __syncthreads();
  }

  if (MODE == 0) {
    // Scatter accumulators into LDS, then coalesced 128-bit stores.
    for (int mt = 0; mt < 2; ++mt)
      for (int nt = 0; nt < 2; ++nt) {
        const int col = wn * 32 + nt * 16 + ln;
        const float bv = bias[n0 + col];
        for (int i = 0; i < 8; ++i)
          Ot[wm * 32 + mt * 16 + hi * 8 + i][col] =
              (_Float16)(acc[mt][nt][i] + bv);
      }
    __syncthreads();
    const int row = tid >> 1;
    const int half = (tid & 1) * 32;
    _Float16* dst = Yh + (size_t)(m0 + row) * Cc + n0 + half;
#pragma unroll
    for (int j = 0; j < 4; ++j)
      *(h8*)(dst + j * 8) = *(const h8*)(&Ot[row][half] + j * 8);
  } else {
    // Accumulator rows map to consecutive l -> 16B vector residual+store.
    for (int mt = 0; mt < 2; ++mt) {
      const int row0 = m0 + wm * 32 + mt * 16 + hi * 8;
      const int bidx = row0 >> 12;
      const int l = row0 & (Ll - 1);
      for (int nt = 0; nt < 2; ++nt) {
        const int col = n0 + wn * 32 + nt * 16 + ln;
        const float bv = bias[col];
        const size_t idx = ((size_t)bidx * Cc + col) * Ll + l;
        f32x4 x0 = *(const f32x4*)(Xres + idx);
        f32x4 x1 = *(const f32x4*)(Xres + idx + 4);
        f32x4 y0, y1;
        for (int i = 0; i < 4; ++i) {
          y0[i] = x0[i] + selu(acc[mt][nt][i] + bv);
          y1[i] = x1[i] + selu(acc[mt][nt][i + 4] + bv);
        }
        *(f32x4*)(Yf + idx) = y0;
        *(f32x4*)(Yf + idx + 4) = y1;
      }
    }
  }
}

// ---------------------------------------------------------------------------
// Flash attention: per block, BM=32 query rows of one batch; key blocks of 32.
// Wave w reduces c-range [w*64, w*64+64) for S and owns O columns
// [w*64, w*64+64). S partials merge via ds_add_f32; online softmax in LDS.
// ---------------------------------------------------------------------------
__global__ __launch_bounds__(256, 1)
void attn_kernel(const _Float16* __restrict__ Q,
                 const _Float16* __restrict__ K,
                 const _Float16* __restrict__ V,
                 _Float16* __restrict__ O) {
  constexpr int BM = 32, BNK = 32;
  __shared__ float S[BM][BNK];
  __shared__ _Float16 P[BM][BNK];
  __shared__ _Float16 Vt[Cc][BNK];  // V^T tile: [c][key]
  __shared__ float rowM[BM], rowL[BM], rowScale[BM];

  const int b = blockIdx.x >> 7;
  const int l0 = (blockIdx.x & 127) * BM;
  const int tid = threadIdx.x;
  const int w = tid >> 5;
  const int lane = tid & 31;
  const int hi = lane >> 4;
  const int ln = lane & 15;

  const _Float16* Qb = Q + (size_t)b * Ll * Cc;
  const _Float16* Kb = K + (size_t)b * Ll * Cc;
  const _Float16* Vb = V + (size_t)b * Ll * Cc;

  v16h qf[2][2];
  for (int mt = 0; mt < 2; ++mt)
    for (int ks = 0; ks < 2; ++ks)
      qf[mt][ks] = load_frag(Qb + (size_t)(l0 + mt * 16) * Cc + w * 64 + ks * 32, Cc);

  v8f oacc[2][4];
  for (int mt = 0; mt < 2; ++mt)
    for (int nt = 0; nt < 4; ++nt)
      oacc[mt][nt] = zero8();

  if (tid < BM) { rowM[tid] = -1e30f; rowL[tid] = 0.f; }
  __syncthreads();

  const float scale = rsqrtf((float)Cc);

  for (int kb = 0; kb < Ll; kb += BNK) {
#pragma unroll
    for (int i = 0; i < 4; ++i) ((float*)S)[tid + i * 256] = 0.f;
    {
      const int key = tid & 31;
      const int cb = (tid >> 5) * 64;
      const _Float16* vp = Vb + (size_t)(kb + key) * Cc + cb;
#pragma unroll
      for (int j = 0; j < 8; ++j) {
        h8 v = *(const h8*)(vp + j * 8);
        for (int e = 0; e < 8; ++e) Vt[cb + j * 8 + e][key] = v[e];
      }
    }
    __syncthreads();

    v8f sp[2][2];
    for (int mt = 0; mt < 2; ++mt)
      for (int nt = 0; nt < 2; ++nt)
        sp[mt][nt] = zero8();
    for (int ks = 0; ks < 2; ++ks) {
      v16h kf[2];
      for (int nt = 0; nt < 2; ++nt)
        kf[nt] = load_frag(Kb + (size_t)(kb + nt * 16) * Cc + w * 64 + ks * 32, Cc);
      for (int mt = 0; mt < 2; ++mt)
        for (int nt = 0; nt < 2; ++nt)
          sp[mt][nt] = wmma_f16(qf[mt][ks], kf[nt], sp[mt][nt]);
    }
    for (int mt = 0; mt < 2; ++mt)
      for (int nt = 0; nt < 2; ++nt)
        for (int i = 0; i < 8; ++i)
          atomicAdd(&S[mt * 16 + hi * 8 + i][nt * 16 + ln], sp[mt][nt][i]);
    __syncthreads();

    if (tid < BM) {
      const int r = tid;
      const float mold = rowM[r];
      float mx = mold;
      for (int c2 = 0; c2 < BNK; ++c2) mx = fmaxf(mx, S[r][c2] * scale);
      const float sc = __expf(mold - mx);
      float sum = 0.f;
      for (int c2 = 0; c2 < BNK; ++c2) {
        const float p = __expf(S[r][c2] * scale - mx);
        sum += p;
        P[r][c2] = (_Float16)p;
      }
      rowM[r] = mx;
      rowL[r] = rowL[r] * sc + sum;
      rowScale[r] = sc;
    }
    __syncthreads();

    v16h pf[2];
    for (int mt = 0; mt < 2; ++mt) pf[mt] = load_frag(&P[mt * 16][0], BNK);
    for (int mt = 0; mt < 2; ++mt)
      for (int nt = 0; nt < 4; ++nt)
        for (int i = 0; i < 8; ++i)
          oacc[mt][nt][i] *= rowScale[mt * 16 + hi * 8 + i];
    for (int nt = 0; nt < 4; ++nt) {
      v16h vf = load_frag(&Vt[w * 64 + nt * 16][0], BNK);
      for (int mt = 0; mt < 2; ++mt)
        oacc[mt][nt] = wmma_f16(pf[mt], vf, oacc[mt][nt]);
    }
    __syncthreads();
  }

  for (int mt = 0; mt < 2; ++mt) {
    for (int nt = 0; nt < 4; ++nt) {
      const int c = w * 64 + nt * 16 + ln;
      for (int i = 0; i < 8; ++i) {
        const int r = mt * 16 + hi * 8 + i;
        const float o = oacc[mt][nt][i] / rowL[r];
        O[(size_t)(b * Ll + l0 + r) * Cc + c] = (_Float16)o;
      }
    }
  }
}

// ---------------------------------------------------------------------------
extern "C" void kernel_launch(void* const* d_in, const int* in_sizes, int n_in,
                              void* d_out, int out_size, void* d_ws, size_t ws_size,
                              hipStream_t stream) {
  const float* x    = (const float*)d_in[0];
  const float* ln_w = (const float*)d_in[1];
  const float* ln_b = (const float*)d_in[2];
  const float* wq   = (const float*)d_in[3];
  const float* bq   = (const float*)d_in[4];
  const float* wk   = (const float*)d_in[5];
  const float* bk   = (const float*)d_in[6];
  const float* wv   = (const float*)d_in[7];
  const float* bv   = (const float*)d_in[8];
  const float* wp   = (const float*)d_in[9];
  const float* bp   = (const float*)d_in[10];
  float* out = (float*)d_out;

  char* ws = (char*)d_ws;
  size_t off = 0;
  auto alloc = [&](size_t bytes) -> void* {
    void* p = ws + off;
    off += (bytes + 255) & ~(size_t)255;
    return p;
  };
  const size_t actsz = (size_t)Mrows * Cc * sizeof(_Float16);
  _Float16* h16  = (_Float16*)alloc(actsz);
  _Float16* q16  = (_Float16*)alloc(actsz);
  _Float16* k16  = (_Float16*)alloc(actsz);
  _Float16* v16  = (_Float16*)alloc(actsz);
  _Float16* o16  = (_Float16*)alloc(actsz);
  _Float16* wq16 = (_Float16*)alloc((size_t)Cc * Cc * sizeof(_Float16));
  _Float16* wk16 = (_Float16*)alloc((size_t)Cc * Cc * sizeof(_Float16));
  _Float16* wv16 = (_Float16*)alloc((size_t)Cc * Cc * sizeof(_Float16));
  _Float16* wp16 = (_Float16*)alloc((size_t)Cc * Cc * sizeof(_Float16));

  const int wcnt = Cc * Cc;
  const int cvtg = (wcnt + 255) / 256;
  cvt_kernel<<<cvtg, 256, 0, stream>>>(wq, wq16, wcnt);
  cvt_kernel<<<cvtg, 256, 0, stream>>>(wk, wk16, wcnt);
  cvt_kernel<<<cvtg, 256, 0, stream>>>(wv, wv16, wcnt);
  cvt_kernel<<<cvtg, 256, 0, stream>>>(wp, wp16, wcnt);

  ln_kernel<<<Bb * (Ll / 32), 256, 0, stream>>>(x, ln_w, ln_b, h16);

  const dim3 gg(Cc / 64, Mrows / 128, 1);
  gemm_kernel<0><<<gg, 256, 0, stream>>>(h16, wq16, bq, q16, nullptr, nullptr);
  gemm_kernel<0><<<gg, 256, 0, stream>>>(h16, wk16, bk, k16, nullptr, nullptr);
  gemm_kernel<0><<<gg, 256, 0, stream>>>(h16, wv16, bv, v16, nullptr, nullptr);

  attn_kernel<<<Bb * (Ll / 32), 256, 0, stream>>>(q16, k16, v16, o16);

  gemm_kernel<1><<<gg, 256, 0, stream>>>(o16, wp16, bp, nullptr, x, out);
}
